// AdaptiveJacobianPrunedViT_90426241450817
// MI455X (gfx1250) — compile-verified, matrix operand order checked by reading, not run on hardware
//
#include <hip/hip_runtime.h>
#include <hip/hip_bf16.h>
#include <cstdint>
#include <cstddef>

// ---------------- model constants ----------------
#define B_        8
#define TMAX      224          // padded token count (7*32, WMMA friendly); real max = 197
#define DIM_      768
#define HEADS_    12
#define HD_       64
#define DEPTH_    12
#define CLASSES_  1000
#define FDIM_     3072
#define QKVDIM_   2304
#define GRID_     14
#define NPATCH_   196
#define SCALE_    0.125f       // 64^-0.5
#define GAMMA_    0.01f
#define MINTOK_   16
#define EPS_      1e-6f

// ---------------- WMMA vector types (wave32, gfx1250) ----------------
typedef __attribute__((ext_vector_type(8)))  __bf16   v8bf;
typedef __attribute__((ext_vector_type(16))) __bf16   v16bf;
typedef __attribute__((ext_vector_type(16))) _Float16 v16h;
typedef __attribute__((ext_vector_type(8)))  float    v8f;

// ISA 16-bit A/B lane layout, K=32: lane = m + 16*g ; element e -> K index:
//   e<8 : K = 8*g + e          (VGPR0..3)
//   e>=8: K = 16 + 8*g + (e-8) (VGPR4..7)
// => per lane-group g the 16 elements are two contiguous 8-float runs at
//    float offsets 8g and 16+8g : perfect for float4 (b128) loads.

static __device__ __forceinline__ v8f wmma_bf16(v16bf a, v16bf b, v8f c) {
  return __builtin_amdgcn_wmma_f32_16x16x32_bf16(false, a, false, b, (short)0, c, false, false);
}
static __device__ __forceinline__ v8f wmma_f16(v16h a, v16h b, v8f c) {
  return __builtin_amdgcn_wmma_f32_16x16x32_f16(false, a, false, b, (short)0, c, false, false);
}

// hardware bf16 converts (v_cvt_pk_bf16_f32) via plain casts
static __device__ __forceinline__ v8bf pack8bf(float4 a, float4 b) {
  v8bf r;
  r[0] = (__bf16)a.x; r[1] = (__bf16)a.y; r[2] = (__bf16)a.z; r[3] = (__bf16)a.w;
  r[4] = (__bf16)b.x; r[5] = (__bf16)b.y; r[6] = (__bf16)b.z; r[7] = (__bf16)b.w;
  return r;
}
static __device__ __forceinline__ v16bf pack16bf(float4 a, float4 b, float4 c, float4 d) {
  v16bf r;
  r[0]  = (__bf16)a.x; r[1]  = (__bf16)a.y; r[2]  = (__bf16)a.z; r[3]  = (__bf16)a.w;
  r[4]  = (__bf16)b.x; r[5]  = (__bf16)b.y; r[6]  = (__bf16)b.z; r[7]  = (__bf16)b.w;
  r[8]  = (__bf16)c.x; r[9]  = (__bf16)c.y; r[10] = (__bf16)c.z; r[11] = (__bf16)c.w;
  r[12] = (__bf16)d.x; r[13] = (__bf16)d.y; r[14] = (__bf16)d.z; r[15] = (__bf16)d.w;
  return r;
}
static __device__ __forceinline__ v16h pack16h(float4 a, float4 b, float4 c, float4 d) {
  v16h r;
  r[0]  = (_Float16)a.x; r[1]  = (_Float16)a.y; r[2]  = (_Float16)a.z; r[3]  = (_Float16)a.w;
  r[4]  = (_Float16)b.x; r[5]  = (_Float16)b.y; r[6]  = (_Float16)b.z; r[7]  = (_Float16)b.w;
  r[8]  = (_Float16)c.x; r[9]  = (_Float16)c.y; r[10] = (_Float16)c.z; r[11] = (_Float16)c.w;
  r[12] = (_Float16)d.x; r[13] = (_Float16)d.y; r[14] = (_Float16)d.z; r[15] = (_Float16)d.w;
  return r;
}

// Load one 16x32-fragment row (16 bf16) for lane-group g from an LDS row
// (row stride 40 bf16 = 80B -> 16B aligned, 16-lane bank-conflict free).
static __device__ __forceinline__ v16bf ldfrag(const __bf16* rowp, int g) {
  v8bf lo = *(const v8bf*)(rowp + 8 * g);
  v8bf hi = *(const v8bf*)(rowp + 16 + 8 * g);
  return __builtin_shufflevector(lo, hi, 0,1,2,3,4,5,6,7,8,9,10,11,12,13,14,15);
}

// ===================================================================
// Generic bf16-WMMA GEMM:  C[m,n] = act( A[m,:] . W[n,:] + bias[n] (+R[m,n]) )
// A:[M,K] row-major, W:[N,K] row-major (i.e. x @ W^T), all fp32 in memory.
// Rows with (m % tokStride) >= *naPtr (pruned/padded tokens) are read as 0
// and stored as exact 0 so padded WMMA tiles stay clean.
// mode: 0 = plain, 1 = exact GELU, 2 = residual add from R.
// ===================================================================
__global__ __launch_bounds__(256)
void gemm_bf16_kernel(const float* __restrict__ A, const float* __restrict__ W,
                      const float* __restrict__ bias, const float* __restrict__ R,
                      float* __restrict__ C, int M, int N, int K,
                      int tokStride, const int* naPtr, int mode) {
  __shared__ __attribute__((aligned(16))) __bf16 As[64][40];
  __shared__ __attribute__((aligned(16))) __bf16 Ws[64][40];
  const int tid  = threadIdx.x;
  const int m0   = blockIdx.x * 64, n0 = blockIdx.y * 64;
  const int Na   = naPtr ? *naPtr : 0x7fffffff;
  const int lr   = tid >> 2;            // load row 0..63
  const int lk   = (tid & 3) * 8;       // load col 0/8/16/24
  const int lane = tid & 31, widx = tid >> 5;
  const int wm   = widx & 3, wn = widx >> 2;     // 4 x 2 wave grid
  const int mr   = lane & 15, g = lane >> 4;
  const int mload = m0 + lr;
  const bool lvalid = (mload < M) && ((mload % tokStride) < Na);
  const float4* Asrc4 = (const float4*)(A + (size_t)mload * K + lk);
  const float4* Wsrc4 = (const float4*)(W + (size_t)(n0 + lr) * K + lk);
  v8f acc0 = {}, acc1 = {};
  for (int k0 = 0; k0 < K; k0 += 32) {
    float4 a0 = make_float4(0.f, 0.f, 0.f, 0.f), a1 = a0;
    if (lvalid) { a0 = Asrc4[k0 >> 2]; a1 = Asrc4[(k0 >> 2) + 1]; }
    float4 w0 = Wsrc4[k0 >> 2], w1 = Wsrc4[(k0 >> 2) + 1];
    if (k0 + 32 < K) __builtin_prefetch(Wsrc4 + (k0 >> 2) + 8, 0, 1);  // global_prefetch_b8
    *(v8bf*)&As[lr][lk] = pack8bf(a0, a1);   // one ds_store_b128
    *(v8bf*)&Ws[lr][lk] = pack8bf(w0, w1);
    __syncthreads();
    v16bf av = ldfrag(&As[wm * 16 + mr][0], g);
    v16bf b0 = ldfrag(&Ws[wn * 32 + mr][0], g);
    v16bf b1 = ldfrag(&Ws[wn * 32 + 16 + mr][0], g);
    acc0 = wmma_bf16(av, b0, acc0);
    acc1 = wmma_bf16(av, b1, acc1);
    __syncthreads();
  }
  const int   n_0 = n0 + wn * 32 + mr;
  const int   n_1 = n_0 + 16;
  const float bb0 = bias ? bias[n_0] : 0.f;
  const float bb1 = bias ? bias[n_1] : 0.f;
#pragma unroll
  for (int r = 0; r < 8; r++) {
    int  m     = m0 + wm * 16 + r + 8 * g;
    bool valid = (m < M) && ((m % tokStride) < Na);
    float v0 = acc0[r] + bb0;
    float v1 = acc1[r] + bb1;
    if (mode == 1) {
      v0 = 0.5f * v0 * (1.f + erff(v0 * 0.70710678118f));
      v1 = 0.5f * v1 * (1.f + erff(v1 * 0.70710678118f));
    } else if (mode == 2) {
      if (valid) { v0 += R[(size_t)m * N + n_0]; v1 += R[(size_t)m * N + n_1]; }
    }
    size_t ro = (size_t)m * N;
    C[ro + n_0] = valid ? v0 : 0.f;
    C[ro + n_1] = valid ? v1 : 0.f;
  }
}

// ---------------- im2col for the stride-16 patch conv ----------------
__global__ void im2col_kernel(const float* __restrict__ x, float* __restrict__ pe) {
  int m = blockIdx.x;                     // 0..B*196-1
  int b = m / NPATCH_, p = m % NPATCH_;
  int gh = p / GRID_, gw = p % GRID_;
  float4* dst = (float4*)(pe + (size_t)m * DIM_);
  for (int j = threadIdx.x; j < DIM_ / 4; j += 256) {
    int k = j * 4;
    int c = k >> 8, rem = k & 255, pp = rem >> 4, qq = rem & 15;
    dst[j] = *(const float4*)&x[(((size_t)(b * 3 + c)) * 224 + gh * 16 + pp) * 224 + gw * 16 + qq];
  }
}

// -------- assemble x0 = [cls ; patches] + pos_embed, zero padding --------
__global__ void assemble_kernel(const float* __restrict__ pe2, const float* __restrict__ cls,
                                const float* __restrict__ pos, float* __restrict__ x) {
  int blk = blockIdx.x;
  int b = blk / TMAX, t = blk % TMAX;
  float4* dst = (float4*)(x + ((size_t)(b * TMAX) + t) * DIM_);
  if (t == 0) {
    const float4* c4 = (const float4*)cls;
    const float4* p4 = (const float4*)pos;
    for (int j = threadIdx.x; j < DIM_ / 4; j += 256) {
      float4 a = c4[j], bb = p4[j];
      dst[j] = make_float4(a.x + bb.x, a.y + bb.y, a.z + bb.z, a.w + bb.w);
    }
  } else if (t < NPATCH_ + 1) {
    const float4* s4 = (const float4*)(pe2 + ((size_t)(b * NPATCH_) + (t - 1)) * DIM_);
    const float4* p4 = (const float4*)(pos + (size_t)t * DIM_);
    for (int j = threadIdx.x; j < DIM_ / 4; j += 256) {
      float4 a = s4[j], bb = p4[j];
      dst[j] = make_float4(a.x + bb.x, a.y + bb.y, a.z + bb.z, a.w + bb.w);
    }
  } else {
    for (int j = threadIdx.x; j < DIM_ / 4; j += 256) dst[j] = make_float4(0.f, 0.f, 0.f, 0.f);
  }
}

// ---------------- LayerNorm: one wave per token row ----------------
__global__ __launch_bounds__(256)
void ln_kernel(const float* __restrict__ x, const float* __restrict__ w,
               const float* __restrict__ bias, float* __restrict__ o, const int* naPtr) {
  int widx = threadIdx.x >> 5, lane = threadIdx.x & 31;
  int idx = blockIdx.x * 8 + widx;        // < B*TMAX
  int t = idx % TMAX;
  int Na = *naPtr;
  const float* xr = x + (size_t)idx * DIM_;
  float* orow = o + (size_t)idx * DIM_;
  if (t >= Na) {
    float4* o4 = (float4*)orow;
    for (int j = lane; j < DIM_ / 4; j += 32) o4[j] = make_float4(0.f, 0.f, 0.f, 0.f);
    return;
  }
  float s = 0.f;
  for (int j = lane; j < DIM_; j += 32) s += xr[j];
  for (int m = 16; m > 0; m >>= 1) s += __shfl_xor(s, m, 32);
  float mu = s * (1.f / DIM_);
  float v = 0.f;
  for (int j = lane; j < DIM_; j += 32) { float d = xr[j] - mu; v += d * d; }
  for (int m = 16; m > 0; m >>= 1) v += __shfl_xor(v, m, 32);
  float rs = rsqrtf(v * (1.f / DIM_) + 1e-6f);
  for (int j = lane; j < DIM_; j += 32) orow[j] = (xr[j] - mu) * rs * w[j] + bias[j];
}

// ---------------- CLS scoring: per (b,h) block -> imp_bh, rho_bh ----------------
__global__ __launch_bounds__(256)
void score_cls_kernel(const float* __restrict__ qkv, float* __restrict__ imp_bh,
                      float* __restrict__ rho_bh, const int* naPtr) {
  __shared__ float qc[HD_];
  __shared__ float la[TMAX];
  __shared__ float red[256];
  int tid = threadIdx.x;
  int bh = blockIdx.x;
  int b = bh / HEADS_, h = bh % HEADS_;
  int Na = *naPtr;
  const float* base = qkv + (size_t)b * TMAX * QKVDIM_;
  if (tid < HD_) qc[tid] = base[h * HD_ + tid];       // q of CLS token
  __syncthreads();
  int t = tid;
  float lg = -3.0e38f;
  if (t < TMAX && t < Na) {
    const float* kr = base + (size_t)t * QKVDIM_ + DIM_ + h * HD_;
    float s = 0.f;
    for (int d = 0; d < HD_; d++) s += qc[d] * kr[d];
    lg = s * SCALE_;
  }
  red[tid] = lg; __syncthreads();
  for (int s = 128; s > 0; s >>= 1) { if (tid < s) red[tid] = fmaxf(red[tid], red[tid + s]); __syncthreads(); }
  float mx = red[0]; __syncthreads();
  float e = (t < TMAX && t < Na) ? expf(lg - mx) : 0.f;
  red[tid] = e; __syncthreads();
  for (int s = 128; s > 0; s >>= 1) { if (tid < s) red[tid] += red[tid + s]; __syncthreads(); }
  float denom = red[0]; __syncthreads();
  float a = e / denom;
  if (t < TMAX) la[t] = a;
  __syncthreads();
  if (t < TMAX) {
    float outv = 0.f;
    if (t >= 1 && t < Na) {
      const float* vr = base + (size_t)t * QKVDIM_ + 2 * DIM_ + h * HD_;
      float s = 0.f;
      for (int d = 0; d < HD_; d++) s += vr[d] * vr[d];
      outv = a * sqrtf(s);
    }
    imp_bh[(size_t)bh * TMAX + t] = outv;
  }
  float partsq = 0.f;
  if (tid < HD_) {
    float oacc = 0.f;
    for (int tt = 0; tt < Na; tt++)
      oacc += la[tt] * base[(size_t)tt * QKVDIM_ + 2 * DIM_ + h * HD_ + tid];
    partsq = oacc * oacc;
  }
  __syncthreads();
  red[tid] = partsq; __syncthreads();
  for (int s = 128; s > 0; s >>= 1) { if (tid < s) red[tid] += red[tid + s]; __syncthreads(); }
  if (tid == 0) rho_bh[bh] = sqrtf(red[0]);
}

// ---------------- prune decision + keep-map (single block, deterministic) ----------------
__global__ __launch_bounds__(256)
void prune_kernel(const float* __restrict__ imp_bh, const float* __restrict__ rho_bh,
                  int* __restrict__ keep_map, int* __restrict__ naPtr,
                  float* __restrict__ prev_mass) {
  __shared__ float imp[TMAX];
  __shared__ float red[256];
  __shared__ int   kept[TMAX];
  __shared__ int   s_nnext;
  __shared__ float s_mass, s_rho;
  int tid = threadIdx.x;
  int Na = naPtr[0];
  int Npatch = Na - 1;
  for (int t = tid; t < TMAX; t += 256) {
    float s = 0.f;
    for (int r = 0; r < B_ * HEADS_; r++) s += imp_bh[(size_t)r * TMAX + t];
    imp[t] = s / (float)(B_ * HEADS_);
  }
  __syncthreads();
  float part = 0.f;
  for (int t = 1 + tid; t < Na; t += 256) part += imp[t];
  red[tid] = part; __syncthreads();
  for (int s = 128; s > 0; s >>= 1) { if (tid < s) red[tid] += red[tid + s]; __syncthreads(); }
  if (tid == 0) s_mass = red[0];
  __syncthreads();
  float rp = 0.f;
  for (int r = tid; r < B_ * HEADS_; r += 256) rp += rho_bh[r];
  red[tid] = rp; __syncthreads();
  for (int s = 128; s > 0; s >>= 1) { if (tid < s) red[tid] += red[tid + s]; __syncthreads(); }
  if (tid == 0) s_rho = red[0] / (float)(B_ * HEADS_);
  __syncthreads();
  if (Npatch > MINTOK_) {                        // block-uniform branch
    if (tid == 0) {
      float mass = s_mass, rho = s_rho, pm = prev_mass[0];
      float kr = 1.f - GAMMA_ * rho / (mass / (pm + EPS_) + EPS_);
      kr = fminf(1.f, fmaxf(0.f, kr));
      int nn = (int)((float)Npatch * kr);
      if (nn < MINTOK_) nn = MINTOK_;
      s_nnext = nn;
      prev_mass[0] = mass;
    }
    __syncthreads();
    int Nnext = s_nnext;
    if (Nnext < Npatch) {                        // block-uniform branch
      for (int t = 1 + tid; t < Na; t += 256) {
        float it = imp[t];
        int c = 0;
        for (int t2 = 1; t2 < Na; t2++) {
          float i2 = imp[t2];
          c += ((i2 > it) || (i2 == it && t2 < t)) ? 1 : 0;
        }
        kept[t] = (c < Nnext) ? 1 : 0;
      }
      __syncthreads();
      for (int t = 1 + tid; t < Na; t += 256) {
        if (kept[t]) {
          int pos = 1;
          for (int t2 = 1; t2 < t; t2++) pos += kept[t2];
          keep_map[pos] = t;
        }
      }
      if (tid == 0) { keep_map[0] = 0; naPtr[0] = Nnext + 1; }
    } else {
      for (int j = tid; j < Na; j += 256) keep_map[j] = j;
    }
  } else {
    for (int j = tid; j < Na; j += 256) keep_map[j] = j;
  }
}

// ---------------- gather kept tokens (compaction) ----------------
__global__ void gather_kernel(const float* __restrict__ xin, float* __restrict__ xout,
                              const int* __restrict__ keep_map, const int* naPtr) {
  int blk = blockIdx.x;
  int b = blk / TMAX, tn = blk % TMAX;
  int Na = *naPtr;
  float4* dst = (float4*)(xout + ((size_t)(b * TMAX) + tn) * DIM_);
  if (tn < Na) {
    int told = keep_map[tn];
    const float4* src = (const float4*)(xin + ((size_t)(b * TMAX) + told) * DIM_);
    for (int j = threadIdx.x; j < DIM_ / 4; j += 256) dst[j] = src[j];
  } else {
    for (int j = threadIdx.x; j < DIM_ / 4; j += 256) dst[j] = make_float4(0.f, 0.f, 0.f, 0.f);
  }
}

// ---------------- attention scores: logits = (Q K^T) * SCALE via bf16 WMMA ----------------
__global__ __launch_bounds__(256)
void attn_score_kernel(const float* __restrict__ qkv, float* __restrict__ logits) {
  int qt = blockIdx.x, h = blockIdx.y, b = blockIdx.z;
  int tid = threadIdx.x, lane = tid & 31, widx = tid >> 5;
  int mr = lane & 15, g = lane >> 4;
  const float4* q4 = (const float4*)(qkv + ((size_t)(b * TMAX) + qt * 16 + mr) * QKVDIM_ + h * HD_);
  v16bf a0 = pack16bf(q4[2*g],     q4[2*g + 1], q4[4 + 2*g],  q4[5 + 2*g]);
  v16bf a1 = pack16bf(q4[8 + 2*g], q4[9 + 2*g], q4[12 + 2*g], q4[13 + 2*g]);
  for (int kt = widx; kt < TMAX / 16; kt += 8) {   // wave-uniform loop
    const float4* k4 = (const float4*)(qkv + ((size_t)(b * TMAX) + kt * 16 + mr) * QKVDIM_ + DIM_ + h * HD_);
    v16bf b0 = pack16bf(k4[2*g],     k4[2*g + 1], k4[4 + 2*g],  k4[5 + 2*g]);
    v16bf b1 = pack16bf(k4[8 + 2*g], k4[9 + 2*g], k4[12 + 2*g], k4[13 + 2*g]);
    v8f acc = {};
    acc = wmma_bf16(a0, b0, acc);
    acc = wmma_bf16(a1, b1, acc);
    float* outp = logits + (((size_t)(b * HEADS_ + h)) * TMAX + qt * 16) * TMAX + kt * 16;
#pragma unroll
    for (int r = 0; r < 8; r++)
      outp[(size_t)(r + 8 * g) * TMAX + mr] = acc[r] * SCALE_;
  }
}

// ---------------- masked softmax: one wave per (b,h,q) row ----------------
__global__ __launch_bounds__(256)
void softmax_kernel(float* __restrict__ logits, const int* naPtr) {
  int widx = threadIdx.x >> 5, lane = threadIdx.x & 31;
  int row = blockIdx.x * 8 + widx;                 // < B*H*TMAX
  float* p = logits + (size_t)row * TMAX;
  int Na = *naPtr;
  int q = row % TMAX;
  if (q >= Na) {
    for (int j = lane; j < TMAX; j += 32) p[j] = 0.f;
    return;
  }
  float mx = -3.0e38f;
  for (int j = lane; j < TMAX; j += 32) if (j < Na) mx = fmaxf(mx, p[j]);
  for (int m = 16; m > 0; m >>= 1) mx = fmaxf(mx, __shfl_xor(mx, m, 32));
  float s = 0.f;
  float vals[TMAX / 32];
  for (int i = 0, j = lane; j < TMAX; j += 32, i++) {
    float e = (j < Na) ? expf(p[j] - mx) : 0.f;
    vals[i] = e; s += e;
  }
  for (int m = 16; m > 0; m >>= 1) s += __shfl_xor(s, m, 32);
  float inv = 1.f / s;
  for (int i = 0, j = lane; j < TMAX; j += 32, i++) p[j] = vals[i] * inv;
}

// ---------------- attn @ V via f16 WMMA (probs in [0,1] -> f16 mantissa wins) ----------------
__global__ __launch_bounds__(128)
void attn_v_kernel(const float* __restrict__ logits, const float* __restrict__ qkv,
                   float* __restrict__ attnout) {
  int qt = blockIdx.x, h = blockIdx.y, b = blockIdx.z;
  int tid = threadIdx.x, lane = tid & 31, widx = tid >> 5;   // 4 waves = 4 d-groups
  int mr = lane & 15, g = lane >> 4;
  int q = qt * 16 + mr;
  const float* arow = logits + (((size_t)(b * HEADS_ + h)) * TMAX + q) * TMAX;
  int d = widx * 16 + mr;
  v8f acc = {};
  for (int kb = 0; kb < TMAX; kb += 32) {
    const float4* a4 = (const float4*)(arow + kb);
    v16h av = pack16h(a4[2*g], a4[2*g + 1], a4[4 + 2*g], a4[5 + 2*g]);
    v16h bv;
#pragma unroll
    for (int e = 0; e < 16; e++) {
      int k = 8 * g + e + ((e >= 8) ? 8 : 0);
      bv[e] = (_Float16)qkv[((size_t)(b * TMAX) + kb + k) * QKVDIM_ + 2 * DIM_ + h * HD_ + d];
    }
    acc = wmma_f16(av, bv, acc);
  }
  float* outp = attnout + ((size_t)(b * TMAX) + qt * 16) * DIM_ + h * HD_ + widx * 16;
#pragma unroll
  for (int r = 0; r < 8; r++)
    outp[(size_t)(r + 8 * g) * DIM_ + mr] = acc[r];
}

// ---------------- classifier head: one wave per (b, class) ----------------
__global__ __launch_bounds__(256)
void head_kernel(const float* __restrict__ xn, const float* __restrict__ hw,
                 const float* __restrict__ hb, float* __restrict__ out) {
  int widx = threadIdx.x >> 5, lane = threadIdx.x & 31;
  int id = blockIdx.x * 8 + widx;                  // 0..B*CLASSES-1
  int b = id / CLASSES_, c = id % CLASSES_;
  const float* xr = xn + (size_t)(b * TMAX) * DIM_;   // token 0 (CLS)
  const float* wr = hw + (size_t)c * DIM_;
  float s = 0.f;
  for (int j = lane; j < DIM_; j += 32) s += xr[j] * wr[j];
  for (int m = 16; m > 0; m >>= 1) s += __shfl_xor(s, m, 32);
  if (lane == 0) out[(size_t)b * CLASSES_ + c] = s + hb[c];
}

__global__ void init_kernel(int* naPtr, float* prev_mass) {
  if (threadIdx.x == 0) { naPtr[0] = NPATCH_ + 1; prev_mass[0] = 1.f; }
}

// ===================================================================
// host-side orchestration
// ===================================================================
static inline int cdiv_(int a, int b) { return (a + b - 1) / b; }

extern "C" void kernel_launch(void* const* d_in, const int* in_sizes, int n_in,
                              void* d_out, int out_size, void* d_ws, size_t ws_size,
                              hipStream_t stream) {
  (void)in_sizes; (void)n_in; (void)out_size; (void)ws_size;
  const float* x_in    = (const float*)d_in[0];
  const float* conv_w  = (const float*)d_in[1];
  const float* conv_b  = (const float*)d_in[2];
  const float* cls_tok = (const float*)d_in[3];
  const float* pos_emb = (const float*)d_in[4];
  const float* ln1_w   = (const float*)d_in[5];
  const float* ln1_b   = (const float*)d_in[6];
  const float* qkv_w   = (const float*)d_in[7];
  const float* qkv_b   = (const float*)d_in[8];
  const float* proj_w  = (const float*)d_in[9];
  const float* proj_b  = (const float*)d_in[10];
  const float* ln2_w   = (const float*)d_in[11];
  const float* ln2_b   = (const float*)d_in[12];
  const float* fc1_w   = (const float*)d_in[13];
  const float* fc1_b   = (const float*)d_in[14];
  const float* fc2_w   = (const float*)d_in[15];
  const float* fc2_b   = (const float*)d_in[16];
  const float* lnf_w   = (const float*)d_in[17];
  const float* lnf_b   = (const float*)d_in[18];
  const float* head_w  = (const float*)d_in[19];
  const float* head_b  = (const float*)d_in[20];
  float* out = (float*)d_out;

  const int MTOK = B_ * TMAX;                // 1792 (28 * 64)
  float* ws = (float*)d_ws;
  size_t off = 0;
  auto alloc = [&](size_t n) { float* p = ws + off; off += (n + 63) & ~(size_t)63; return p; };
  float* xcur    = alloc((size_t)MTOK * DIM_);
  float* xg      = alloc((size_t)MTOK * DIM_);
  float* xo      = alloc((size_t)MTOK * DIM_);
  float* xn      = alloc((size_t)MTOK * DIM_);
  float* attnout = alloc((size_t)MTOK * DIM_);
  float* qkvbuf  = alloc((size_t)MTOK * QKVDIM_);
  float* hbuf    = alloc((size_t)MTOK * FDIM_);
  float* logits  = alloc((size_t)B_ * HEADS_ * TMAX * TMAX);
  float* pebuf   = alloc((size_t)1600 * DIM_);      // padded to 25*64 rows
  float* pebuf2  = alloc((size_t)1600 * DIM_);
  float* imp_bh  = alloc((size_t)B_ * HEADS_ * TMAX);
  float* rho_bh  = alloc(64 + B_ * HEADS_);         // padded
  float* pmass   = alloc(64);
  int*   keepmap = (int*)alloc(256);
  int*   naPtr   = (int*)alloc(64);

  auto gemm = [&](const float* A, const float* W, const float* bias, const float* R,
                  float* C, int M, int N, int K, int tokStride, const int* na, int mode) {
    dim3 g(cdiv_(M, 64), N / 64);
    gemm_bf16_kernel<<<g, dim3(256), 0, stream>>>(A, W, bias, R, C, M, N, K, tokStride, na, mode);
  };
  auto ln = [&](const float* x, const float* w, const float* b, float* o) {
    ln_kernel<<<MTOK / 8, 256, 0, stream>>>(x, w, b, o, naPtr);
  };

  init_kernel<<<1, 32, 0, stream>>>(naPtr, pmass);
  // patch embed: im2col + WMMA GEMM, then assemble tokens (+cls, +pos, zero pad)
  im2col_kernel<<<B_ * NPATCH_, 256, 0, stream>>>(x_in, pebuf);
  gemm(pebuf, conv_w, conv_b, nullptr, pebuf2, B_ * NPATCH_, DIM_, DIM_, 1 << 30, nullptr, 0);
  assemble_kernel<<<MTOK, 256, 0, stream>>>(pebuf2, cls_tok, pos_emb, xcur);

  for (int i = 0; i < DEPTH_; i++) {
    const float* l1w = ln1_w + (size_t)i * DIM_;
    const float* l1b = ln1_b + (size_t)i * DIM_;
    const float* qw  = qkv_w + (size_t)i * QKVDIM_ * DIM_;
    const float* qb  = qkv_b + (size_t)i * QKVDIM_;
    // --- scoring pass on current token set (as in reference) ---
    ln(xcur, l1w, l1b, xn);
    gemm(xn, qw, qb, nullptr, qkvbuf, MTOK, QKVDIM_, DIM_, TMAX, naPtr, 0);
    score_cls_kernel<<<B_ * HEADS_, 256, 0, stream>>>(qkvbuf, imp_bh, rho_bh, naPtr);
    prune_kernel<<<1, 256, 0, stream>>>(imp_bh, rho_bh, keepmap, naPtr, pmass);
    gather_kernel<<<MTOK, 256, 0, stream>>>(xcur, xg, keepmap, naPtr);
    // --- block on (possibly pruned) tokens ---
    ln(xg, l1w, l1b, xn);
    gemm(xn, qw, qb, nullptr, qkvbuf, MTOK, QKVDIM_, DIM_, TMAX, naPtr, 0);
    attn_score_kernel<<<dim3(TMAX / 16, HEADS_, B_), 256, 0, stream>>>(qkvbuf, logits);
    softmax_kernel<<<B_ * HEADS_ * TMAX / 8, 256, 0, stream>>>(logits, naPtr);
    attn_v_kernel<<<dim3(TMAX / 16, HEADS_, B_), 128, 0, stream>>>(logits, qkvbuf, attnout);
    gemm(attnout, proj_w + (size_t)i * DIM_ * DIM_, proj_b + (size_t)i * DIM_,
         xg, xo, MTOK, DIM_, DIM_, TMAX, naPtr, 2);
    ln(xo, ln2_w + (size_t)i * DIM_, ln2_b + (size_t)i * DIM_, xn);
    gemm(xn, fc1_w + (size_t)i * FDIM_ * DIM_, fc1_b + (size_t)i * FDIM_,
         nullptr, hbuf, MTOK, FDIM_, DIM_, TMAX, naPtr, 1);
    gemm(hbuf, fc2_w + (size_t)i * DIM_ * FDIM_, fc2_b + (size_t)i * DIM_,
         xo, xcur, MTOK, DIM_, FDIM_, TMAX, naPtr, 2);
  }
  ln(xcur, lnf_w, lnf_b, xn);
  head_kernel<<<CLASSES_, 256, 0, stream>>>(xn, head_w, head_b, out);
}
